// OnlineLRU_41807211660055
// MI455X (gfx1250) — matrix operand order repaired and checked by reading, not compiled, not run on hardware
//
#include <hip/hip_runtime.h>
#include <hip/hip_bf16.h>

#define T_STEPS 512
#define D_HID   512
#define IN_DIM  256

typedef __attribute__((ext_vector_type(16))) __bf16 v16bf;
typedef __attribute__((ext_vector_type(8)))  float  v8f;
typedef __attribute__((ext_vector_type(2)))  float  f2;
typedef __attribute__((ext_vector_type(4)))  float  f4;

union V16BF { v16bf v; __bf16 e[16]; };
union F8    { v8f  v; float  e[8]; };

// Fragment buffer geometry: 32 tiles x 8 k-chunks x 32 lanes x 16 bf16
#define FRAG_ELEMS (32 * 8 * 32 * 16)   // 131072 bf16 = 256 KB

// ---------------------------------------------------------------------------
// Kernel 0: split f32 -> (hi, lo) bf16 and pre-swizzle into WMMA fragment
// layout, so the GEMM inner loop is pure fragment-load + v_wmma.
//   which 0: A fragments from X  (16x32 A layout, ISA 7.12.2)
//   which 1: B fragments from Br (32x16 B layout)
//   which 2: B fragments from Bi
// ---------------------------------------------------------------------------
__global__ __launch_bounds__(256) void lru_split_swizzle(
    const float* __restrict__ X, const float* __restrict__ Br,
    const float* __restrict__ Bi,
    __bf16* __restrict__ XfH,  __bf16* __restrict__ XfL,
    __bf16* __restrict__ BrfH, __bf16* __restrict__ BrfL,
    __bf16* __restrict__ BifH, __bf16* __restrict__ BifL)
{
    const int gid   = blockIdx.x * 256 + threadIdx.x;  // 0..24575
    const int which = gid >> 13;                       // 3 x 8192 fragments
    const int fid   = gid & 8191;                      // (tile*8 + c)*32 + lane
    const int lane  = fid & 31;
    const int c     = (fid >> 5) & 7;
    const int tile  = fid >> 8;
    const int half  = lane >> 4;
    const int mn    = lane & 15;

    const float* src;
    __bf16 *dh, *dl;
    if (which == 0)      { src = X;  dh = XfH;  dl = XfL;  }
    else if (which == 1) { src = Br; dh = BrfH; dl = BrfL; }
    else                 { src = Bi; dh = BifH; dl = BifL; }

    const float* row = src + (tile * 16 + mn) * IN_DIM + c * 32;
    const bool isA = (which == 0);

    V16BF hi, lo;
#pragma unroll
    for (int j2 = 0; j2 < 16; ++j2) {
        const int j = j2 >> 1, pos = j2 & 1;
        int K;
        if (isA) {             // A 16x32: v0-3 K=khA+2j, v4-7 K=16+khA+2(j-4)
            const int khA = half * 8;
            K = (j < 4) ? (khA + 2 * j + pos) : (16 + khA + 2 * (j - 4) + pos);
        } else {               // B 32x16: v0-3 K=khB+2j, v4-7 K=16+khB+2(j-4)
            const int khB = half * 8;
            K = (j < 4) ? (khB + 2 * j + pos) : (16 + khB + 2 * (j - 4) + pos);
        }
        const float v  = row[K];
        const __bf16 vh = (__bf16)v;
        hi.e[j2] = vh;
        lo.e[j2] = (__bf16)(v - (float)vh);
    }
    ((v16bf*)dh)[fid] = hi.v;
    ((v16bf*)dl)[fid] = lo.v;
}

// ---------------------------------------------------------------------------
// Kernel A: Yr[t,d] = sum_i X[t,i]*Br[d,i];  Yi[t,d] = sum_i X[t,i]*Bi[d,i]
// One wave per 16x16 (t,d) tile. Inner loop: 6 fragment loads + 6 WMMAs
// (hi*hi + lo*hi + hi*lo compensation for ~f32 accuracy), f32 accumulate.
// ---------------------------------------------------------------------------
__global__ __launch_bounds__(256) void lru_gemm_wmma(
    const __bf16* __restrict__ XfH,  const __bf16* __restrict__ XfL,
    const __bf16* __restrict__ BrfH, const __bf16* __restrict__ BrfL,
    const __bf16* __restrict__ BifH, const __bf16* __restrict__ BifL,
    float* __restrict__ Yr, float* __restrict__ Yi)
{
    const int lane = threadIdx.x & 31;
    const int wave = threadIdx.x >> 5;
    const int tile = blockIdx.x * 8 + wave;   // 0..1023 (32x32 tiles)
    const int tT   = tile >> 5;               // t-tile
    const int tD   = tile & 31;               // d-tile

    v8f accR = {};
    v8f accI = {};

#pragma unroll
    for (int c = 0; c < 8; ++c) {
        const int ai = (tT * 8 + c) * 32 + lane;
        const int bi = (tD * 8 + c) * 32 + lane;
        const v16bf ah  = ((const v16bf*)XfH)[ai];
        const v16bf al  = ((const v16bf*)XfL)[ai];
        const v16bf brh = ((const v16bf*)BrfH)[bi];
        const v16bf brl = ((const v16bf*)BrfL)[bi];
        const v16bf bih = ((const v16bf*)BifH)[bi];
        const v16bf bil = ((const v16bf*)BifL)[bi];

        accR = __builtin_amdgcn_wmma_f32_16x16x32_bf16(false, ah, false, brh, (short)0, accR, false, false);
        accR = __builtin_amdgcn_wmma_f32_16x16x32_bf16(false, al, false, brh, (short)0, accR, false, false);
        accR = __builtin_amdgcn_wmma_f32_16x16x32_bf16(false, ah, false, brl, (short)0, accR, false, false);
        accI = __builtin_amdgcn_wmma_f32_16x16x32_bf16(false, ah, false, bih, (short)0, accI, false, false);
        accI = __builtin_amdgcn_wmma_f32_16x16x32_bf16(false, al, false, bih, (short)0, accI, false, false);
        accI = __builtin_amdgcn_wmma_f32_16x16x32_bf16(false, ah, false, bil, (short)0, accI, false, false);
    }

    // C/D layout: VGPR j -> row M = j + (lane>>4)*8, col N = lane&15
    const int m      = lane & 15;
    const int hiHalf = lane >> 4;
    const int t0     = tT * 16;
    const int d0     = tD * 16;
    F8 outR, outI;
    outR.v = accR;
    outI.v = accI;
#pragma unroll
    for (int j = 0; j < 8; ++j) {
        const int trow = t0 + hiHalf * 8 + j;
        Yr[trow * D_HID + d0 + m] = outR.e[j];
        Yi[trow * D_HID + d0 + m] = outI.e[j];
    }
}

// ---------------------------------------------------------------------------
// Kernel B: diagonal scans for h, gl, gg over T steps. One thread per d.
//   u2 = Y[t,d] (= B@x),  u1 = exp(gamma_log[d]) * u2 (= B_norm@x)
//   h  <- Lam*h  + u1 ; gl <- Lam*gl + h_old ; gg <- Lam*gg + u2
// ---------------------------------------------------------------------------
__global__ __launch_bounds__(256) void lru_scan_diag(
    const float* __restrict__ Yr, const float* __restrict__ Yi,
    const float* __restrict__ h0, const float* __restrict__ gl0,
    const float* __restrict__ gg0,
    const float* __restrict__ nu_log, const float* __restrict__ theta_log,
    const float* __restrict__ gamma_log,
    float* __restrict__ out)
{
    const int d = blockIdx.x * blockDim.x + threadIdx.x;
    if (d >= D_HID) return;

    const float mag = __expf(-__expf(nu_log[d]));
    const float ang = __expf(theta_log[d]);
    const float lr  = mag * __cosf(ang);
    const float li  = mag * __sinf(ang);
    const float g   = __expf(gamma_log[d]);

    float hr = h0[2 * d],  hi = h0[2 * d + 1];
    float glr = gl0[2 * d], gli = gl0[2 * d + 1];
    float ggr = gg0[2 * d], ggi = gg0[2 * d + 1];

    f2* hout  = (f2*)out;
    f2* glout = (f2*)(out + 2 * T_STEPS * D_HID);
    f2* ggout = (f2*)(out + 4 * T_STEPS * D_HID);

#pragma unroll 4
    for (int t = 0; t < T_STEPS; ++t) {
        const float u2r = Yr[t * D_HID + d];
        const float u2i = Yi[t * D_HID + d];
        const float hro = hr, hio = hi;
        hr = lr * hro - li * hio + g * u2r;
        hi = lr * hio + li * hro + g * u2i;
        const float nglr = lr * glr - li * gli + hro;
        const float ngli = lr * gli + li * glr + hio;
        glr = nglr; gli = ngli;
        const float nggr = lr * ggr - li * ggi + u2r;
        const float nggi = lr * ggi + li * ggr + u2i;
        ggr = nggr; ggi = nggi;

        const int idx = t * D_HID + d;
        f2 vh;  vh.x = hr;  vh.y = hi;
        f2 vl;  vl.x = glr; vl.y = gli;
        f2 vg;  vg.x = ggr; vg.y = ggi;
        hout[idx]  = vh;
        glout[idx] = vl;
        ggout[idx] = vg;
    }
}

// ---------------------------------------------------------------------------
// Kernel C: gB recurrence + streaming output (512 MB -> B128 NT stores).
//   gB[t,d,i] = Lam[d]*gB[t-1,d,i] + gamma_log[d]*x[t,i]
// One block per d (Lam uniform -> SGPRs); 128 threads, 2 complex elems each.
// ---------------------------------------------------------------------------
__global__ __launch_bounds__(128) void lru_gB_stream(
    const float* __restrict__ X,    // [T, I]
    const float* __restrict__ gB0,  // [D, I] complex (interleaved)
    const float* __restrict__ nu_log, const float* __restrict__ theta_log,
    const float* __restrict__ gamma_log,
    float* __restrict__ gBout)      // [T, D, I] complex (interleaved)
{
    const int d   = blockIdx.x;   // uniform -> scalar loads for Lam, glog
    const int tid = threadIdx.x;  // 0..127, covers i0 = 2*tid, i0+1

    const float mag  = __expf(-__expf(nu_log[d]));
    const float ang  = __expf(theta_log[d]);
    const float lr   = mag * __cosf(ang);
    const float li   = mag * __sinf(ang);
    const float glog = gamma_log[d];

    const int i0 = tid * 2;
    float gr0 = gB0[2 * (d * IN_DIM + i0)];
    float gi0 = gB0[2 * (d * IN_DIM + i0) + 1];
    float gr1 = gB0[2 * (d * IN_DIM + i0 + 1)];
    float gi1 = gB0[2 * (d * IN_DIM + i0 + 1) + 1];

    const f2* Xv  = (const f2*)X;  // pairs of x along i
    f4* outv      = (f4*)gBout;    // 2 complex elems per store

#pragma unroll 4
    for (int t = 0; t < T_STEPS; ++t) {
        const f2 x = Xv[t * (IN_DIM / 2) + tid];
        const float ngr0 = lr * gr0 - li * gi0 + glog * x.x;  // added term real
        const float ngi0 = lr * gi0 + li * gr0;
        const float ngr1 = lr * gr1 - li * gi1 + glog * x.y;
        const float ngi1 = lr * gi1 + li * gr1;
        gr0 = ngr0; gi0 = ngi0; gr1 = ngr1; gi1 = ngi1;
        f4 v; v.x = gr0; v.y = gi0; v.z = gr1; v.w = gi1;
        __builtin_nontemporal_store(
            v, outv + ((size_t)t * D_HID + d) * (IN_DIM / 2) + tid);
    }
}

// ---------------------------------------------------------------------------
extern "C" void kernel_launch(void* const* d_in, const int* in_sizes, int n_in,
                              void* d_out, int out_size, void* d_ws, size_t ws_size,
                              hipStream_t stream) {
    const float* X   = (const float*)d_in[0];  // inputs   [T, I]
    const float* h0  = (const float*)d_in[1];  // complex  [D]
    const float* gl0 = (const float*)d_in[2];  // complex  [D]
    const float* gg0 = (const float*)d_in[3];  // complex  [D]
    const float* gB0 = (const float*)d_in[4];  // complex  [D, I]
    const float* nu  = (const float*)d_in[5];  // [D]
    const float* th  = (const float*)d_in[6];  // [D]
    const float* gm  = (const float*)d_in[7];  // [D]
    const float* Br  = (const float*)d_in[8];  // [D, I]
    const float* Bi  = (const float*)d_in[9];  // [D, I]

    float* out = (float*)d_out;

    // Workspace layout (3.5 MB):
    float*  Yr   = (float*)d_ws;               // [T, D]   1 MB
    float*  Yi   = Yr + T_STEPS * D_HID;       // [T, D]   1 MB
    __bf16* XfH  = (__bf16*)(Yi + T_STEPS * D_HID);
    __bf16* XfL  = XfH  + FRAG_ELEMS;
    __bf16* BrfH = XfL  + FRAG_ELEMS;
    __bf16* BrfL = BrfH + FRAG_ELEMS;
    __bf16* BifH = BrfL + FRAG_ELEMS;
    __bf16* BifL = BifH + FRAG_ELEMS;

    // 0: split + fragment-swizzle (3 * 8192 fragments)
    lru_split_swizzle<<<96, 256, 0, stream>>>(X, Br, Bi, XfH, XfL,
                                              BrfH, BrfL, BifH, BifL);
    // A: batched matvecs as WMMA GEMM (1024 tiles, 8 waves/block)
    lru_gemm_wmma<<<128, 256, 0, stream>>>(XfH, XfL, BrfH, BrfL, BifH, BifL,
                                           Yr, Yi);
    // B: diagonal scans -> h_seq, gl_seq, gg_seq
    lru_scan_diag<<<(D_HID + 255) / 256, 256, 0, stream>>>(
        Yr, Yi, h0, gl0, gg0, nu, th, gm, out);
    // C: gB_seq streaming recurrence (dominant 512 MB output, B128 NT stores)
    lru_gB_stream<<<D_HID, 128, 0, stream>>>(
        X, gB0, nu, th, gm, out + 6 * T_STEPS * D_HID);
}
